// MPSelfAttention_9405978378557
// MI455X (gfx1250) — matrix operand order repaired and checked by reading, not compiled, not run on hardware
//
#include <hip/hip_runtime.h>

#define DEVINL __device__ __forceinline__

typedef __bf16 bf16_t;
typedef __attribute__((ext_vector_type(16))) __bf16 v16bf;
typedef __attribute__((ext_vector_type(8)))  __bf16 v8bf;
typedef __attribute__((ext_vector_type(8)))  float  v8f;

static constexpr int Bz = 2, Sz = 2048, Dz = 1024, Hz = 16, Dh = 64;
static constexpr float EPSF = 1e-6f;

// ---------------------------------------------------------------------------
// WMMA helpers (CDNA5 wave32, 16x16x32 bf16 -> f32)
// ---------------------------------------------------------------------------
DEVINL v8f wmma_bf16(v16bf a, v16bf b, v8f c) {
  return __builtin_amdgcn_wmma_f32_16x16x32_bf16(false, a, false, b, (short)0, c,
                                                 false, false);
}

// A-fragment: 16x32 (MxK) bf16 from row-major [row0.., k0..k0+31], leading dim ld.
// Lanes 0-15: rows M=lane, K=k0..k0+15 (contig 32B);
// lanes 16-31: K=k0+16..k0+31.
DEVINL v16bf load_frag_a(const bf16_t* base, int ld, int row0, int k0, int lane) {
  const int r  = lane & 15;
  const int hk = (lane >> 4) << 4;  // 0 or 16
  return *(const v16bf*)(base + (long)(row0 + r) * ld + k0 + hk);
}

// B-fragment: 32x16 (KxN) bf16 where B[k][n] = W[col0+n][k0+k], W row-major
// (transposed operand). Lanes 0-15: K=k0..k0+15 (contig 32B); lanes 16-31:
// K=k0+16..k0+31.  N = lane%16.
DEVINL v16bf load_frag_b(const bf16_t* base, int ld, int col0, int k0, int lane) {
  const int n  = lane & 15;
  const int hk = (lane >> 4) << 4;  // 0 or 16
  return *(const v16bf*)(base + (long)(col0 + n) * ld + k0 + hk);
}

// ---------------------------------------------------------------------------
// Async global->LDS staging (CDNA5 TDM-adjacent path, tracked by ASYNCcnt)
// ---------------------------------------------------------------------------
typedef __attribute__((address_space(3))) char lds_char;

DEVINL unsigned lds_off(const void* p) {
  // generic -> LDS addrspace(3) -> raw 32-bit LDS byte offset
  return (unsigned)(unsigned long long)(const lds_char*)p;
}

DEVINL void async_g2l_b128(unsigned lds_byte_off, const void* gsrc) {
  asm volatile("global_load_async_to_lds_b128 %0, %1, off"
               :: "v"(lds_byte_off), "v"((unsigned long long)(size_t)gsrc)
               : "memory");
}

DEVINL void wait_async_le1() { asm volatile("s_wait_asynccnt 0x1" ::: "memory"); }
DEVINL void wait_async_le0() { asm volatile("s_wait_asynccnt 0x0" ::: "memory"); }

// Stage a 32x32 bf16 A chunk (rows m0.., cols k0..k0+31) into LDS tile.
// Threads 0..127 each move one 16-byte piece (wave-uniform predicate).
DEVINL void stage_a_async(const bf16_t* A, int K, int m0, int k0,
                          bf16_t* tile, int tid) {
  if (tid < 128) {
    const int row   = tid >> 2;
    const int piece = tid & 3;
    const bf16_t* g = A + (long)(m0 + row) * K + k0 + piece * 8;
    async_g2l_b128(lds_off(tile + row * 32 + piece * 8), g);
  }
}

// ---------------------------------------------------------------------------
// f32 -> bf16 convert
// ---------------------------------------------------------------------------
__global__ void __launch_bounds__(256)
f32_to_bf16_kernel(const float* __restrict__ in, bf16_t* __restrict__ out, long n) {
  long i = (long)blockIdx.x * blockDim.x + threadIdx.x;
  if (i < n) out[i] = (bf16_t)in[i];
}

// ---------------------------------------------------------------------------
// Row L2-normalize (scale = rsqrt(sum(v^2)*inv_div + eps)) f32 -> bf16.
// inv_div = 1 -> unit_norm (sum); inv_div = 1/cols -> rms_norm (mean)
// ---------------------------------------------------------------------------
__global__ void __launch_bounds__(256)
row_norm_to_bf16(const float* __restrict__ in, bf16_t* __restrict__ out,
                 int cols, float inv_div) {
  const long row = blockIdx.x;
  const float* rp = in + row * cols;
  float s = 0.f;
  for (int i = threadIdx.x; i < cols; i += 256) { float v = rp[i]; s += v * v; }
#pragma unroll
  for (int sh = 16; sh >= 1; sh >>= 1) s += __shfl_xor(s, sh, 32);
  __shared__ float red[8];
  if ((threadIdx.x & 31) == 0) red[threadIdx.x >> 5] = s;
  __syncthreads();
  float tot = 0.f;
#pragma unroll
  for (int w = 0; w < 8; ++w) tot += red[w];
  const float sc = rsqrtf(tot * inv_div + EPSF);
  bf16_t* op = out + row * cols;
  for (int i = threadIdx.x; i < cols; i += 256) op[i] = (bf16_t)(rp[i] * sc);
}

// ---------------------------------------------------------------------------
// GEMM: C(MxN) = A(MxK) * B(NxK)^T, bf16 in, f32 accumulate.
// Block = 8 waves sharing 32 M-rows, covering 256 N-cols (8 x 32-wide tiles).
// A K-chunk (32x32) is double-buffered in LDS via async global->LDS loads;
// B-fragments stream straight from global into the WMMA operands.
// Requires: M % 32 == 0, N % 256 == 0, K % 32 == 0.
// ---------------------------------------------------------------------------
template <bool OUT_BF16>
__global__ void __launch_bounds__(256)
wmma_gemm_nt(const bf16_t* __restrict__ A, const bf16_t* __restrict__ Bm,
             void* __restrict__ Cout, int M, int N, int K) {
  __shared__ __align__(16) bf16_t atile[2][32 * 32];
  const int tid  = threadIdx.x;
  const int lane = tid & 31;
  const int wave = tid >> 5;
  const int ngrp = N >> 8;                 // N / 256
  const int m0 = (blockIdx.x / ngrp) * 32;
  const int n0 = ((blockIdx.x % ngrp) * 8 + wave) * 32;

  v8f acc00 = {}, acc01 = {}, acc10 = {}, acc11 = {};
  stage_a_async(A, K, m0, 0, atile[0], tid);
  int cur = 0;
  for (int k0 = 0; k0 < K; k0 += 32) {
    const bool has_next = (k0 + 32 < K);
    if (has_next) {
      stage_a_async(A, K, m0, k0 + 32, atile[cur ^ 1], tid);
      __builtin_prefetch(Bm + (long)(n0 + (lane & 15)) * K + k0 + 32, 0, 1);
      wait_async_le1();                    // previous chunk landed (in-order)
    } else {
      wait_async_le0();
    }
    __syncthreads();                       // publish LDS chunk to all waves

    v16bf a0 = load_frag_a(atile[cur], 32, 0, 0, lane);
    v16bf a1 = load_frag_a(atile[cur], 32, 16, 0, lane);
    v16bf b0 = load_frag_b(Bm, K, n0, k0, lane);
    v16bf b1 = load_frag_b(Bm, K, n0 + 16, k0, lane);
    acc00 = wmma_bf16(a0, b0, acc00);
    acc01 = wmma_bf16(a0, b1, acc01);
    acc10 = wmma_bf16(a1, b0, acc10);
    acc11 = wmma_bf16(a1, b1, acc11);

    __syncthreads();                       // all reads done before re-stage
    cur ^= 1;
  }

  const int cn = lane & 15, half = lane >> 4;
#pragma unroll
  for (int i = 0; i < 2; ++i)
#pragma unroll
    for (int j = 0; j < 2; ++j) {
      const v8f& acc = (i == 0) ? (j == 0 ? acc00 : acc01)
                                : (j == 0 ? acc10 : acc11);
#pragma unroll
      for (int r = 0; r < 8; ++r) {
        const long row = m0 + 16 * i + r + 8 * half;
        const long col = n0 + 16 * j + cn;
        if constexpr (OUT_BF16)
          ((bf16_t*)Cout)[row * N + col] = (bf16_t)acc[r];
        else
          ((float*)Cout)[row * N + col] = acc[r];
      }
    }
}

// ---------------------------------------------------------------------------
// RoPE + per-head RMS on q,k; split qkv; store V transposed (B,H,Dh,S).
// One wave per (b,s,h). qkv rows: [q(0:D) | k(D:2D) | v(2D:3D)], head h at h*Dh.
// ---------------------------------------------------------------------------
__global__ void __launch_bounds__(256)
rope_split(const bf16_t* __restrict__ qkv, const float* __restrict__ freqs,
           bf16_t* __restrict__ q, bf16_t* __restrict__ k,
           bf16_t* __restrict__ vT) {
  const int lane = threadIdx.x & 31;
  const long w = (long)blockIdx.x * 8 + (threadIdx.x >> 5);  // (b*S+s)*H + h
  const int h = (int)(w % Hz);
  const long bs = w / Hz;            // b*S + s
  const int s = (int)(bs % Sz);
  const long b = bs / Sz;
  const long bhh = b * Hz + h;
  const long bhs = bhh * Sz + s;
  const bf16_t* row = qkv + bs * (3 * Dz) + h * Dh;

  float sn, cs;
  const float f = freqs[(long)s * (Dh / 2) + lane];
  __sincosf(f, &sn, &cs);

#pragma unroll
  for (int t = 0; t < 2; ++t) {  // t=0: q, t=1: k
    const bf16_t* src = row + t * Dz;
    const float x1 = (float)src[lane];
    const float x2 = (float)src[lane + 32];
    const float y1 = x1 * cs - x2 * sn;
    const float y2 = x1 * sn + x2 * cs;
    float ss = y1 * y1 + y2 * y2;
#pragma unroll
    for (int sh = 16; sh >= 1; sh >>= 1) ss += __shfl_xor(ss, sh, 32);
    const float sc = rsqrtf(ss * (1.0f / Dh) + EPSF);
    bf16_t* dst = (t == 0 ? q : k) + bhs * Dh;
    dst[lane]      = (bf16_t)(y1 * sc);
    dst[lane + 32] = (bf16_t)(y2 * sc);
  }
  const bf16_t* vsrc = row + 2 * Dz;
#pragma unroll
  for (int d = lane; d < Dh; d += 32)
    vT[(bhh * Dh + d) * (long)Sz + s] = vsrc[d];
}

// ---------------------------------------------------------------------------
// Causal flash attention. grid=(S/128, B*H), block=256 (8 waves).
// Each wave owns a 16-query tile; loops over 32-key blocks with online softmax.
// QK^T: 4 WMMAs/block; P transposed C-layout->A-layout through wave-private LDS;
// P*V: 4 WMMAs/block against pre-transposed V.
// ---------------------------------------------------------------------------
__global__ void __launch_bounds__(256)
flash_attn(const bf16_t* __restrict__ q, const bf16_t* __restrict__ k,
           const bf16_t* __restrict__ vT, float* __restrict__ out) {
  __shared__ __align__(64) bf16_t pbuf[8][16 * 32];
  const int lane = threadIdx.x & 31;
  const int wave = threadIdx.x >> 5;
  const int bh = blockIdx.y;  // b*H + h
  const int b = bh / Hz, h = bh % Hz;
  const int q0 = (blockIdx.x * 8 + wave) * 16;
  const bf16_t* qh = q + (long)bh * Sz * Dh;
  const bf16_t* kh = k + (long)bh * Sz * Dh;
  const bf16_t* vh = vT + (long)bh * Dh * Sz;

  const v16bf qa0 = load_frag_a(qh, Dh, q0, 0, lane);
  const v16bf qa1 = load_frag_a(qh, Dh, q0, 32, lane);

  float mrun[8], lrun[8];
  v8f o0 = {}, o1 = {}, o2 = {}, o3 = {};
#pragma unroll
  for (int r = 0; r < 8; ++r) { mrun[r] = -3e38f; lrun[r] = 0.f; }
  const int cn = lane & 15, half = lane >> 4;
  bf16_t* pb = &pbuf[wave][0];

  for (int n0 = 0; n0 < q0 + 16; n0 += 32) {
    v8f s0 = {}, s1 = {};
    s0 = wmma_bf16(qa0, load_frag_b(kh, Dh, n0, 0, lane), s0);
    s0 = wmma_bf16(qa1, load_frag_b(kh, Dh, n0, 32, lane), s0);
    s1 = wmma_bf16(qa0, load_frag_b(kh, Dh, n0 + 16, 0, lane), s1);
    s1 = wmma_bf16(qa1, load_frag_b(kh, Dh, n0 + 16, 32, lane), s1);

#pragma unroll
    for (int r = 0; r < 8; ++r) {
      const int row = q0 + r + 8 * half;
      float v0 = (n0 + cn <= row)      ? s0[r] * 0.125f : -3e38f;
      float v1 = (n0 + 16 + cn <= row) ? s1[r] * 0.125f : -3e38f;
      float mx = fmaxf(v0, v1);
#pragma unroll
      for (int sh = 8; sh >= 1; sh >>= 1) mx = fmaxf(mx, __shfl_xor(mx, sh, 32));
      const float mnew = fmaxf(mrun[r], mx);
      const float sc = __expf(mrun[r] - mnew);
      const float p0 = __expf(v0 - mnew);
      const float p1 = __expf(v1 - mnew);
      float rs = p0 + p1;
#pragma unroll
      for (int sh = 8; sh >= 1; sh >>= 1) rs += __shfl_xor(rs, sh, 32);
      lrun[r] = lrun[r] * sc + rs;
      mrun[r] = mnew;
      o0[r] *= sc; o1[r] *= sc; o2[r] *= sc; o3[r] *= sc;
      const int m = r + 8 * half;
      pb[m * 32 + cn]      = (bf16_t)p0;   // wave-private LDS tile, no barrier
      pb[m * 32 + 16 + cn] = (bf16_t)p1;   // needed (DS in-order within wave)
    }
    const v16bf pa = load_frag_a(pb, 32, 0, 0, lane);
    o0 = wmma_bf16(pa, load_frag_b(vh, Sz, 0,  n0, lane), o0);
    o1 = wmma_bf16(pa, load_frag_b(vh, Sz, 16, n0, lane), o1);
    o2 = wmma_bf16(pa, load_frag_b(vh, Sz, 32, n0, lane), o2);
    o3 = wmma_bf16(pa, load_frag_b(vh, Sz, 48, n0, lane), o3);
  }

#pragma unroll
  for (int r = 0; r < 8; ++r) {
    const float inv = 1.f / lrun[r];
    const long row = (long)b * Sz + q0 + r + 8 * half;
    float* op = out + row * Dz + h * Dh + cn;
    op[0]  = o0[r] * inv;
    op[16] = o1[r] * inv;
    op[32] = o2[r] * inv;
    op[48] = o3[r] * inv;
  }
}

// ---------------------------------------------------------------------------
// Host launch
// ---------------------------------------------------------------------------
extern "C" void kernel_launch(void* const* d_in, const int* in_sizes, int n_in,
                              void* d_out, int out_size, void* d_ws, size_t ws_size,
                              hipStream_t stream) {
  (void)in_sizes; (void)n_in; (void)out_size; (void)ws_size;
  const float* x     = (const float*)d_in[0];
  const float* rpf   = (const float*)d_in[1];
  const float* w_in  = (const float*)d_in[2];
  const float* w_out = (const float*)d_in[3];

  char* ws = (char*)d_ws;
  size_t off = 0;
  auto alloc = [&](size_t bytes) -> void* {
    off = (off + 255) & ~(size_t)255;
    void* p = ws + off;
    off += bytes;
    return p;
  };

  bf16_t* xb  = (bf16_t*)alloc((size_t)Bz * Sz * Dz * 2);        // x bf16
  bf16_t* wib = (bf16_t*)alloc((size_t)3 * Dz * Dz * 2);         // unit_norm(w_in)
  bf16_t* wob = (bf16_t*)alloc((size_t)Dz * Dz * 2);             // unit_norm(w_out)
  bf16_t* qb  = (bf16_t*)alloc((size_t)Bz * Hz * Sz * Dh * 2);   // q (B,H,S,Dh)
  bf16_t* kb  = (bf16_t*)alloc((size_t)Bz * Hz * Sz * Dh * 2);   // k (B,H,S,Dh)
  bf16_t* vtb = (bf16_t*)alloc((size_t)Bz * Hz * Dh * Sz * 2);   // vT (B,H,Dh,S)
  // qkv bf16 region (25.2MB) is consumed by rope_split, then reused:
  //   [0 .. 16.8MB)  attention output f32 (B,S,D)
  //   [16.8 .. 25.2) rms-normalized activations bf16 (B,S,D)
  char* shared_reg = (char*)alloc((size_t)Bz * Sz * 3 * Dz * 2);
  bf16_t* qkvb = (bf16_t*)shared_reg;
  float*  attn = (float*)shared_reg;
  bf16_t* ab   = (bf16_t*)(shared_reg + (size_t)Bz * Sz * Dz * 4);

  const long nX = (long)Bz * Sz * Dz;
  f32_to_bf16_kernel<<<(int)((nX + 255) / 256), 256, 0, stream>>>(x, xb, nX);
  row_norm_to_bf16<<<3 * Dz, 256, 0, stream>>>(w_in, wib, Dz, 1.0f);
  row_norm_to_bf16<<<Dz, 256, 0, stream>>>(w_out, wob, Dz, 1.0f);

  {  // qkv = xb @ wib^T : (4096 x 1024) x (3072 x 1024)^T -> bf16
    const int M = Bz * Sz, N = 3 * Dz, K = Dz;
    const int blocks = (M / 32) * (N / 256);
    wmma_gemm_nt<true><<<blocks, 256, 0, stream>>>(xb, wib, qkvb, M, N, K);
  }

  rope_split<<<(Bz * Sz * Hz) / 8, 256, 0, stream>>>(qkvb, rpf, qb, kb, vtb);

  {
    dim3 grid(Sz / 128, Bz * Hz);
    flash_attn<<<grid, 256, 0, stream>>>(qb, kb, vtb, attn);
  }

  row_norm_to_bf16<<<Bz * Sz, 256, 0, stream>>>(attn, ab, Dz, 1.0f / Dz);

  {  // out = ab @ wob^T -> f32
    const int M = Bz * Sz, N = Dz, K = Dz;
    const int blocks = (M / 32) * (N / 256);
    wmma_gemm_nt<false><<<blocks, 256, 0, stream>>>(ab, wob, d_out, M, N, K);
  }
}
